// MultihopAttention_10608569221509
// MI455X (gfx1250) — matrix operand order repaired and checked by reading, not compiled, not run on hardware
//
#include <hip/hip_runtime.h>
#include <hip/hip_bf16.h>

#define B_   32
#define S_   512
#define H_   768
#define K_   4
#define HOP_ 3
#define MS_  (B_ * S_)   // 16384
#define H2_  (2 * H_)    // 1536
#define H3_  (3 * H_)    // 2304

typedef __bf16 bf16_t;
typedef __attribute__((ext_vector_type(16))) __bf16 v16bf;
typedef __attribute__((ext_vector_type(8)))  __bf16 v8bf;
typedef __attribute__((ext_vector_type(8)))  float  v8f;
typedef int v4i_ __attribute__((vector_size(16)));   // 128-bit payload type for async-LDS builtin

#define AS1 __attribute__((address_space(1)))
#define AS3 __attribute__((address_space(3)))

#if __has_builtin(__builtin_amdgcn_global_load_async_to_lds_b128)
#define HAVE_ASYNC_LDS 1
#else
#define HAVE_ASYNC_LDS 0
#endif

__device__ __forceinline__ void wait_async_lds()
{
#if HAVE_ASYNC_LDS
#if __has_builtin(__builtin_amdgcn_s_wait_asynccnt)
    __builtin_amdgcn_s_wait_asynccnt(0);
#else
    asm volatile("s_wait_asynccnt 0" ::: "memory");
#endif
#endif
}

__device__ __forceinline__ float sigmoidf_(float x) { return 1.f / (1.f + __expf(-x)); }

// ---------------------------------------------------------------------------
// Generic batched WMMA GEMM:  C[z] = A[z] (MxKd, bf16) * Bt[z]^T + bias[z]
// Bt is stored pre-transposed as [N x Kd] so B fetches are contiguous.
//
// Block = 256 threads = 8 waves; block tile 256(M) x 64(N).
// Each wave computes a 32x64 strip: 2 M-tiles x 4 N-tiles = 8 accumulators,
// so 8 WMMAs amortize 2 A-fragment loads + one shared B panel.
// The 64x32 bf16 B panel (4KB) is staged once per block into LDS,
// double-buffered, via GLOBAL_LOAD_ASYNC_TO_LDS_B128 when available.
//
// Fragment layouts per CDNA5 ISA §7.12.2:
//   A (16x32 bf16): lane lo holds row m; e<8 -> k=e+hi*8, e>=8 -> k=e+8+hi*8
//   B (32x16 bf16): lane lo holds column n; e<8 -> k=e+hi*8, e>=8 -> k=e+8+hi*8 (same as A)
//   C/D (16x16 f32): VGPR i -> m = i + hi*8, n = lo
// ---------------------------------------------------------------------------
#define LDSROW_B 112   // bytes per staged B row (64 used + pad for bank spread)
#define BUFSZ_B  (64 * LDSROW_B)

template <bool OUT_BF16, bool HAS_BIAS>
__global__ void gemm_wmma_bf16(const bf16_t* __restrict__ A,
                               const bf16_t* __restrict__ Bt,
                               void* __restrict__ Cv,
                               const float* __restrict__ bias,
                               int M, int N, int Kd,
                               long long sA, long long sB, long long sC, long long sBias)
{
    __shared__ __align__(16) unsigned char sBpanel[2][BUFSZ_B];

    const int tid  = threadIdx.x;
    const int lane = tid & 31;
    const int wave = tid >> 5;
    const int lo   = lane & 15;
    const int hi   = (lane >> 4) & 1;

    const int m0 = blockIdx.y * 256 + wave * 32;
    const int n0 = blockIdx.x * 64;
    const bool active = (m0 < M);   // inactive waves still stage B + barrier

    const bf16_t* Ab = A  + (long long)blockIdx.z * sA;
    const bf16_t* Bb = Bt + (long long)blockIdx.z * sB;

    // staging assignment: thread tid copies one 16B chunk of the 4KB panel
    const int srow = tid >> 2;   // 0..63  (panel row = local n)
    const int scol = tid & 3;    // 0..3   (16B chunk within 64B row)

    auto stage = [&](int kb, int buf) {
        const bf16_t* src = Bb + (long long)(n0 + srow) * Kd + kb + scol * 8;
        unsigned char* dst = &sBpanel[buf][srow * LDSROW_B + scol * 16];
#if HAVE_ASYNC_LDS
        __builtin_amdgcn_global_load_async_to_lds_b128(
            (AS1 v4i_*)(v4i_*)(void*)const_cast<bf16_t*>(src),
            (AS3 v4i_*)(v4i_*)(void*)dst, 0, 0);
#else
        *(v8bf*)dst = *(const v8bf*)src;
#endif
    };

    v8f acc[2][4] = {};

    const long long ar0 = (long long)(m0 + lo) * Kd;        // M-tile 0 row
    const long long ar1 = ar0 + (long long)16 * Kd;         // M-tile 1 row

    stage(0, 0);
    int buf = 0;

    for (int kb = 0; kb < Kd; kb += 32) {
        wait_async_lds();
        __syncthreads();
        if (kb + 32 < Kd) stage(kb + 32, buf ^ 1);

        if (active) {
            // A fragments: 2 M-tiles, two contiguous 16B chunks each
            v8bf x0 = *(const v8bf*)(Ab + ar0 + kb + hi * 8);
            v8bf x1 = *(const v8bf*)(Ab + ar0 + kb + 16 + hi * 8);
            v8bf y0 = *(const v8bf*)(Ab + ar1 + kb + hi * 8);
            v8bf y1 = *(const v8bf*)(Ab + ar1 + kb + 16 + hi * 8);
            v16bf a0, a1;
#pragma unroll
            for (int e = 0; e < 8; ++e) {
                a0[e] = x0[e]; a0[e + 8] = x1[e];
                a1[e] = y0[e]; a1[e + 8] = y1[e];
            }
            __builtin_prefetch((const void*)(Ab + ar0 + kb + 128), 0, 1);
            __builtin_prefetch((const void*)(Ab + ar1 + kb + 128), 0, 1);

#pragma unroll
            for (int j = 0; j < 4; ++j) {
                const unsigned char* bp =
                    &sBpanel[buf][(j * 16 + lo) * LDSROW_B + hi * 16];
                v8bf b0 = *(const v8bf*)(bp);
                v8bf b1 = *(const v8bf*)(bp + 32);
                v16bf bb;
#pragma unroll
                for (int e = 0; e < 8; ++e) { bb[e] = b0[e]; bb[e + 8] = b1[e]; }
                acc[0][j] = __builtin_amdgcn_wmma_f32_16x16x32_bf16(
                    false, a0, false, bb, (short)0, acc[0][j], false, false);
                acc[1][j] = __builtin_amdgcn_wmma_f32_16x16x32_bf16(
                    false, a1, false, bb, (short)0, acc[1][j], false, false);
            }
        }
        buf ^= 1;
    }

    if (!active) return;

    const long long cbase = (long long)blockIdx.z * sC;
#pragma unroll
    for (int t = 0; t < 2; ++t) {
#pragma unroll
        for (int j = 0; j < 4; ++j) {
            const int n = n0 + j * 16 + lo;
            float bv = 0.f;
            if (HAS_BIAS) bv = bias[(long long)blockIdx.z * sBias + n];
#pragma unroll
            for (int i = 0; i < 8; ++i) {
                const int m = m0 + t * 16 + i + hi * 8;
                const float v = acc[t][j][i] + bv;
                if (OUT_BF16)
                    ((bf16_t*)Cv)[cbase + (long long)m * N + n] = (bf16_t)v;
                else
                    ((float*)Cv)[cbase + (long long)m * N + n] = v;
            }
        }
    }
}

// ---------------------------------------------------------------------------
// Elementwise f32 -> bf16 conversion
// ---------------------------------------------------------------------------
__global__ void cvt_f32_to_bf16(const float* __restrict__ src, bf16_t* __restrict__ dst, int n)
{
    int i = blockIdx.x * blockDim.x + threadIdx.x;
    if (i < n) dst[i] = (bf16_t)src[i];
}

// ---------------------------------------------------------------------------
// LDS-tiled transpose + convert: src f32 [R,C] -> dst bf16 [C,R], batched.
// ---------------------------------------------------------------------------
__global__ void transpose_cvt_f32_bf16(const float* __restrict__ src, bf16_t* __restrict__ dst,
                                       int R, int C, long long sSrc, long long sDst)
{
    __shared__ float tile[32][33];
    const float* s = src + (long long)blockIdx.z * sSrc;
    bf16_t*      d = dst + (long long)blockIdx.z * sDst;
    const int c0 = blockIdx.x * 32, r0 = blockIdx.y * 32;

    for (int i = threadIdx.y; i < 32; i += 8) {
        int r = r0 + i, c = c0 + threadIdx.x;
        tile[i][threadIdx.x] = (r < R && c < C) ? s[(long long)r * C + c] : 0.f;
    }
    __syncthreads();
    for (int i = threadIdx.y; i < 32; i += 8) {
        int c = c0 + i, r = r0 + threadIdx.x;
        if (c < C && r < R) d[(long long)c * R + r] = (bf16_t)tile[threadIdx.x][i];
    }
}

// ---------------------------------------------------------------------------
// cb[k,j] = Ws_b[k,j] + sum_h pw_b[k,h] * Ws_W[k,h,j]
// ---------------------------------------------------------------------------
__global__ void combined_bias(const float* __restrict__ pwb, const float* __restrict__ WsW,
                              const float* __restrict__ Wsb, float* __restrict__ cb)
{
    __shared__ float sp[H_];
    const int k = blockIdx.x;
    for (int i = threadIdx.x; i < H_; i += blockDim.x) sp[i] = pwb[k * H_ + i];
    __syncthreads();
    for (int j = threadIdx.x; j < H_; j += blockDim.x) {
        float acc = Wsb[k * H_ + j];
        const float* col = WsW + (long long)k * H_ * H_ + j;
        for (int h = 0; h < H_; ++h) acc += sp[h] * col[(long long)h * H_];
        cb[k * H_ + j] = acc;
    }
}

// ---------------------------------------------------------------------------
// Init: h = 0, h_bf = 0, q_bf = bf16(question)
// ---------------------------------------------------------------------------
__global__ void init_state(const float* __restrict__ question, float* __restrict__ h,
                           bf16_t* __restrict__ h_bf, bf16_t* __restrict__ q_bf, int n)
{
    int i = blockIdx.x * blockDim.x + threadIdx.x;
    if (i < n) { h[i] = 0.f; h_bf[i] = (bf16_t)0.f; q_bf[i] = (bf16_t)question[i]; }
}

// ---------------------------------------------------------------------------
// Per (k,b): scores[s] = We_b[k] + sum_h We[k,h]*tanh(Whs_bf[k,b,s,h] + Wht[k,b,h])
// then softmax over s (S=512) -> alpha[k,b,s].  grid = K*B blocks, 256 threads.
// Whs is read in contiguous 16B bf16 chunks (global_load_b128, L2-resident).
// ---------------------------------------------------------------------------
__global__ void attn_scores_softmax(const bf16_t* __restrict__ Whs, const float* __restrict__ Wht,
                                    const float* __restrict__ WeW, const float* __restrict__ Web,
                                    float* __restrict__ alpha)
{
    __shared__ float sWht[H_];
    __shared__ float sWe[H_];
    __shared__ float sc[S_];
    __shared__ float red[256];

    const int kb = blockIdx.x;
    const int k = kb / B_, b = kb % B_;
    const int tid = threadIdx.x;
    const int wave = tid >> 5, lane = tid & 31;

    for (int i = tid; i < H_; i += 256) {
        sWht[i] = Wht[((long long)k * B_ + b) * H_ + i];
        sWe[i]  = WeW[k * H_ + i];
    }
    __syncthreads();

    const float web = Web[k];
    const bf16_t* base = Whs + ((long long)k * MS_ + (long long)b * S_) * H_;

    for (int s = wave; s < S_; s += 8) {
        const bf16_t* row = base + (long long)s * H_;
        float acc = 0.f;
        for (int c = lane; c < H_ / 8; c += 32) {   // 96 chunks of 8 bf16 (16B)
            v8bf v = *(const v8bf*)(row + c * 8);
#pragma unroll
            for (int e = 0; e < 8; ++e) {
                const int h = c * 8 + e;
                acc += sWe[h] * tanhf((float)v[e] + sWht[h]);
            }
        }
        for (int off = 16; off > 0; off >>= 1) acc += __shfl_down(acc, off, 32);
        if (lane == 0) sc[s] = acc + web;
    }
    __syncthreads();

    // softmax over S in LDS
    float m = -1e30f;
    for (int s = tid; s < S_; s += 256) m = fmaxf(m, sc[s]);
    red[tid] = m; __syncthreads();
    for (int st = 128; st > 0; st >>= 1) {
        if (tid < st) red[tid] = fmaxf(red[tid], red[tid + st]);
        __syncthreads();
    }
    m = red[0]; __syncthreads();

    float sum = 0.f;
    for (int s = tid; s < S_; s += 256) {
        float e = __expf(sc[s] - m);
        sc[s] = e; sum += e;
    }
    red[tid] = sum; __syncthreads();
    for (int st = 128; st > 0; st >>= 1) {
        if (tid < st) red[tid] += red[tid + st];
        __syncthreads();
    }
    const float inv = 1.f / red[0];
    float* out = alpha + ((long long)k * B_ + b) * S_;
    for (int s = tid; s < S_; s += 256) out[s] = sc[s] * inv;
}

// ---------------------------------------------------------------------------
// Per b: alpha_linear, L2-normalized prob (stored at hop 0), argmax gather,
// build x_bf = concat(bf(question), bf(text[b, idx])).  grid=B, block=S.
// ---------------------------------------------------------------------------
__global__ void head_mix_gather(const float* __restrict__ alpha, const float* __restrict__ headwW,
                                const float* __restrict__ headwb, const float* __restrict__ text,
                                const bf16_t* __restrict__ q_bf, float* __restrict__ out_prob,
                                bf16_t* __restrict__ x_bf, int store_prob)
{
    __shared__ float rf[S_];
    __shared__ int   ri[S_];

    const int b = blockIdx.x;
    const int s = threadIdx.x;

    float al = headwb[0];
#pragma unroll
    for (int k = 0; k < K_; ++k)
        al += alpha[((long long)k * B_ + b) * S_ + s] * headwW[k];

    // sum of squares -> scale
    rf[s] = al * al; __syncthreads();
    for (int st = S_ / 2; st > 0; st >>= 1) {
        if (s < st) rf[s] += rf[s + st];
        __syncthreads();
    }
    const float scale = sqrtf(rf[0]);
    __syncthreads();

    if (store_prob) out_prob[(long long)b * S_ + s] = al / scale;

    // argmax (first-max tie-break via strict >)
    rf[s] = al; ri[s] = s; __syncthreads();
    for (int st = S_ / 2; st > 0; st >>= 1) {
        if (s < st) {
            if (rf[s + st] > rf[s]) { rf[s] = rf[s + st]; ri[s] = ri[s + st]; }
        }
        __syncthreads();
    }
    const int idx = ri[0];

    // x = concat(question, text[b, idx]) in bf16
    bf16_t* xr = x_bf + (long long)b * H2_;
    const float* trow = text + ((long long)b * S_ + idx) * H_;
    for (int h = s; h < H_; h += S_) {
        xr[h]      = q_bf[(long long)b * H_ + h];
        xr[H_ + h] = (bf16_t)trow[h];
    }
}

// ---------------------------------------------------------------------------
// GRU pointwise: consumes gx=[B,3H], gh=[B,3H]; updates h, h_bf, q_bf,
// writes h_states slot into d_out.
// ---------------------------------------------------------------------------
__global__ void gru_pointwise(const float* __restrict__ gx, const float* __restrict__ gh,
                              float* __restrict__ h, bf16_t* __restrict__ h_bf,
                              bf16_t* __restrict__ q_bf, float* __restrict__ out, int hop)
{
    int i = blockIdx.x * blockDim.x + threadIdx.x;
    if (i >= B_ * H_) return;
    const int b = i / H_, j = i % H_;
    const float* gxr = gx + (long long)b * H3_;
    const float* ghr = gh + (long long)b * H3_;
    const float r = sigmoidf_(gxr[j] + ghr[j]);
    const float z = sigmoidf_(gxr[H_ + j] + ghr[H_ + j]);
    const float n = tanhf(gxr[2 * H_ + j] + r * ghr[2 * H_ + j]);
    const float hp = h[i];
    const float hn = (1.f - z) * n + z * hp;
    h[i] = hn;
    h_bf[i] = (bf16_t)hn;
    q_bf[i] = (bf16_t)hn;
    out[(long long)B_ * S_ + ((long long)b * HOP_ + hop) * H_ + j] = hn;
}

// ---------------------------------------------------------------------------
extern "C" void kernel_launch(void* const* d_in, const int* in_sizes, int n_in,
                              void* d_out, int out_size, void* d_ws, size_t ws_size,
                              hipStream_t stream)
{
    const float* question = (const float*)d_in[0];
    const float* text     = (const float*)d_in[1];
    const float* pw_W     = (const float*)d_in[2];
    const float* pw_b     = (const float*)d_in[3];
    const float* Ws_W     = (const float*)d_in[4];
    const float* Ws_b     = (const float*)d_in[5];
    const float* Wt_W     = (const float*)d_in[6];
    const float* Wt_b     = (const float*)d_in[7];
    const float* We_W     = (const float*)d_in[8];
    const float* We_b     = (const float*)d_in[9];
    const float* headw_W  = (const float*)d_in[10];
    const float* headw_b  = (const float*)d_in[11];
    const float* gru_Wih  = (const float*)d_in[12];
    const float* gru_Whh  = (const float*)d_in[13];
    const float* gru_bih  = (const float*)d_in[14];
    const float* gru_bhh  = (const float*)d_in[15];
    float* out = (float*)d_out;

    // bump allocator over d_ws
    char* wptr = (char*)d_ws;
    auto alloc = [&](size_t bytes) -> void* {
        void* p = (void*)wptr;
        wptr += (bytes + 255) & ~(size_t)255;
        return p;
    };

    bf16_t* text_bf = (bf16_t*)alloc((size_t)MS_ * H_ * sizeof(bf16_t));
    bf16_t* pw_bf   = (bf16_t*)alloc((size_t)K_ * H_ * H_ * sizeof(bf16_t));
    bf16_t* WsT_bf  = (bf16_t*)alloc((size_t)K_ * H_ * H_ * sizeof(bf16_t));
    bf16_t* WtT_bf  = (bf16_t*)alloc((size_t)K_ * H_ * H_ * sizeof(bf16_t));
    float*  CW_f32  = (float*)alloc((size_t)K_ * H_ * H_ * sizeof(float));
    bf16_t* CWT_bf  = (bf16_t*)alloc((size_t)K_ * H_ * H_ * sizeof(bf16_t));
    float*  cb      = (float*)alloc((size_t)K_ * H_ * sizeof(float));
    bf16_t* WihT_bf = (bf16_t*)alloc((size_t)H3_ * H2_ * sizeof(bf16_t));
    bf16_t* WhhT_bf = (bf16_t*)alloc((size_t)H3_ * H_ * sizeof(bf16_t));
    bf16_t* Whs_bf  = (bf16_t*)alloc((size_t)K_ * MS_ * H_ * sizeof(bf16_t));  // 96 MiB, L2-resident
    float*  Wht     = (float*)alloc((size_t)K_ * B_ * H_ * sizeof(float));
    float*  alpha   = (float*)alloc((size_t)K_ * B_ * S_ * sizeof(float));
    bf16_t* q_bf    = (bf16_t*)alloc((size_t)B_ * H_ * sizeof(bf16_t));
    bf16_t* h_bf    = (bf16_t*)alloc((size_t)B_ * H_ * sizeof(bf16_t));
    float*  h_f32   = (float*)alloc((size_t)B_ * H_ * sizeof(float));
    bf16_t* x_bf    = (bf16_t*)alloc((size_t)B_ * H2_ * sizeof(bf16_t));
    float*  gates_x = (float*)alloc((size_t)B_ * H3_ * sizeof(float));
    float*  gates_h = (float*)alloc((size_t)B_ * H3_ * sizeof(float));

    // ---- one-time prep -----------------------------------------------------
    {
        int n = MS_ * H_;
        cvt_f32_to_bf16<<<(n + 255) / 256, 256, 0, stream>>>(text, text_bf, n);
    }
    {
        int n = K_ * H_ * H_;
        cvt_f32_to_bf16<<<(n + 255) / 256, 256, 0, stream>>>(pw_W, pw_bf, n);
    }
    transpose_cvt_f32_bf16<<<dim3(24, 24, K_), dim3(32, 8), 0, stream>>>(
        Ws_W, WsT_bf, H_, H_, (long long)H_ * H_, (long long)H_ * H_);
    transpose_cvt_f32_bf16<<<dim3(24, 24, K_), dim3(32, 8), 0, stream>>>(
        Wt_W, WtT_bf, H_, H_, (long long)H_ * H_, (long long)H_ * H_);
    transpose_cvt_f32_bf16<<<dim3(H3_ / 32, H2_ / 32, 1), dim3(32, 8), 0, stream>>>(
        gru_Wih, WihT_bf, H2_, H3_, 0, 0);
    transpose_cvt_f32_bf16<<<dim3(H3_ / 32, H_ / 32, 1), dim3(32, 8), 0, stream>>>(
        gru_Whh, WhhT_bf, H_, H3_, 0, 0);
    combined_bias<<<K_, 256, 0, stream>>>(pw_b, Ws_W, Ws_b, cb);

    // CW[k] = pw_W[k] @ Ws_W[k]   (f32 out)
    gemm_wmma_bf16<false, false><<<dim3(H_ / 64, 3, K_), 256, 0, stream>>>(
        pw_bf, WsT_bf, CW_f32, nullptr, H_, H_, H_,
        (long long)H_ * H_, (long long)H_ * H_, (long long)H_ * H_, 0);
    transpose_cvt_f32_bf16<<<dim3(24, 24, K_), dim3(32, 8), 0, stream>>>(
        CW_f32, CWT_bf, H_, H_, (long long)H_ * H_, (long long)H_ * H_);

    // W_hs[k] = text @ CW[k] + cb[k]   (bf16 out, K x 16384 x 768)
    gemm_wmma_bf16<true, true><<<dim3(H_ / 64, MS_ / 256, K_), 256, 0, stream>>>(
        text_bf, CWT_bf, Whs_bf, cb, MS_, H_, H_,
        0, (long long)H_ * H_, (long long)MS_ * H_, (long long)H_);

    init_state<<<(B_ * H_ + 255) / 256, 256, 0, stream>>>(question, h_f32, h_bf, q_bf, B_ * H_);

    // ---- hop loop ----------------------------------------------------------
    for (int hop = 0; hop < HOP_; ++hop) {
        // W_ht[k] = q @ Wt_W[k] + Wt_b[k]   (f32, K x 32 x 768)
        gemm_wmma_bf16<false, true><<<dim3(H_ / 64, 1, K_), 256, 0, stream>>>(
            q_bf, WtT_bf, Wht, Wt_b, B_, H_, H_,
            0, (long long)H_ * H_, (long long)B_ * H_, (long long)H_);

        attn_scores_softmax<<<K_ * B_, 256, 0, stream>>>(Whs_bf, Wht, We_W, We_b, alpha);

        head_mix_gather<<<B_, S_, 0, stream>>>(alpha, headw_W, headw_b, text, q_bf,
                                               out, x_bf, hop == 0 ? 1 : 0);

        // gx = x @ Wih + bih   (32 x 2304, Kd=1536)
        gemm_wmma_bf16<false, true><<<dim3(H3_ / 64, 1, 1), 256, 0, stream>>>(
            x_bf, WihT_bf, gates_x, gru_bih, B_, H3_, H2_, 0, 0, 0, 0);
        // gh = h @ Whh + bhh   (32 x 2304, Kd=768)
        gemm_wmma_bf16<false, true><<<dim3(H3_ / 64, 1, 1), 256, 0, stream>>>(
            h_bf, WhhT_bf, gates_h, gru_bhh, B_, H3_, H_, 0, 0, 0, 0);

        gru_pointwise<<<(B_ * H_ + 255) / 256, 256, 0, stream>>>(
            gates_x, gates_h, h_f32, h_bf, q_bf, out, hop);
    }

    (void)in_sizes; (void)n_in; (void)out_size; (void)ws_size;
}
